// GNNPolicy_17154099380396
// MI455X (gfx1250) — matrix operand order repaired and checked by reading, not compiled, not run on hardware
//
#include <hip/hip_runtime.h>
#include <hip/hip_bf16.h>
#include <math.h>

// ---------------- constants ----------------
#define K_NCONS  50000
#define K_NVARS  100000
#define K_NEDGES 1000000
#define K_EPS    1e-5f

// ---------------- WMMA plumbing ----------------
typedef __attribute__((ext_vector_type(16))) __bf16 v16bf_t;
typedef __attribute__((ext_vector_type(8)))  float  v8f_t;

union Frag { v16bf_t v; unsigned int u[8]; };

__device__ __forceinline__ unsigned short f2bf(float f) {
  unsigned int u = __float_as_uint(f);
  u += 0x7FFFu + ((u >> 16) & 1u);           // round-to-nearest-even
  return (unsigned short)(u >> 16);
}

__device__ __forceinline__ v8f_t wmma_bf16(const Frag& a, const Frag& b, v8f_t c) {
  return __builtin_amdgcn_wmma_f32_16x16x32_bf16(false, a.v, false, b.v, (short)0, c,
                                                 false, false);
}

// A fragment (16x32 bf16, ISA 7.12.2): lane m=l&15, kh=l>>4;
// elements 0..7  -> K = k0 + 8*kh + e
// elements 8..15 -> K = k0 + 16 + 8*kh + (e-8)
__device__ __forceinline__ void load_a_frag(Frag& a, const unsigned short* sA,
                                            int stride, int k0, int lane) {
  const int m = lane & 15, kh = lane >> 4;
  const unsigned short* p = sA + m * stride + k0 + 8 * kh;
  *reinterpret_cast<uint4*>(&a.u[0]) = *reinterpret_cast<const uint4*>(p);
  *reinterpret_cast<uint4*>(&a.u[4]) = *reinterpret_cast<const uint4*>(p + 16);
}

// B fragment: packed layout P[((kb*4+nb)*32+lane)*16 + e] (16 contiguous bf16/lane)
__device__ __forceinline__ void load_b_frag(Frag& b, const unsigned short* P,
                                            int kb, int nb, int lane) {
  const unsigned short* p = P + (((size_t)kb * 4 + nb) * 32 + (size_t)lane) * 16;
  *reinterpret_cast<uint4*>(&b.u[0]) = *reinterpret_cast<const uint4*>(p);
  *reinterpret_cast<uint4*>(&b.u[4]) = *reinterpret_cast<const uint4*>(p + 8);
}

__device__ __forceinline__ v8f_t gemm_tile(const unsigned short* sA, int stride,
                                           const unsigned short* P, int nb, int KB,
                                           int lane, v8f_t acc) {
  for (int kb = 0; kb < KB; ++kb) {
    Frag a, b;
    load_a_frag(a, sA, stride, kb * 32, lane);
    load_b_frag(b, P, kb, nb, lane);
    acc = wmma_bf16(a, b, acc);
  }
  return acc;
}

// ---------------- weight packing ----------------
// Pack f32 W[K][64] into B-fragment layout (K zero-padded to Kpad, mult of 32).
// B K-map per lane: K = kb*32 + 8*kh + e (e<8), kb*32 + 16 + 8*kh + (e-8) (e>=8).
__global__ void pack_w_kernel(const float* __restrict__ W, int K, int Kpad,
                              unsigned short* __restrict__ P) {
  int idx = blockIdx.x * blockDim.x + threadIdx.x;
  int total = (Kpad >> 5) * 2048;
  if (idx >= total) return;
  int e    = idx & 15;
  int lane = (idx >> 4) & 31;
  int nb   = (idx >> 9) & 3;
  int kb   = idx >> 11;
  int n  = nb * 16 + (lane & 15);
  int kh = lane >> 4;
  int k  = kb * 32 + (e < 8 ? kh * 8 + e : 16 + kh * 8 + (e - 8));
  float v = (k < K) ? W[(size_t)k * 64 + n] : 0.f;
  P[idx] = f2bf(v);
}

// ---------------- misc kernels ----------------
__global__ void zero_kernel(float* __restrict__ p, long long n) {
  long long i = (long long)blockIdx.x * blockDim.x + threadIdx.x;
  if (i < n) p[i] = 0.f;
}

// Periodic embeddings: [N,F] -> relu(cos/sin(2*pi*coef*x) @ W + b) -> [N, F*24]
__global__ void embed_kernel(const float* __restrict__ x, int N, int F,
                             const float* __restrict__ coef,   // [F,16]
                             const float* __restrict__ W,      // [32,24]
                             const float* __restrict__ b,      // [24]
                             float* __restrict__ out) {
  int idx = blockIdx.x * blockDim.x + threadIdx.x;
  if (idx >= N * F) return;
  int n = idx / F, f = idx - n * F;
  float xv = x[(size_t)n * F + f] * 6.283185307179586f;
  float fv[32];
#pragma unroll
  for (int k = 0; k < 16; ++k) {
    float s, c;
    __sincosf(xv * coef[f * 16 + k], &s, &c);
    fv[k] = c; fv[16 + k] = s;
  }
  float* op = out + (size_t)n * (F * 24) + f * 24;
  for (int d = 0; d < 24; ++d) {
    float acc = b[d];
#pragma unroll
    for (int k = 0; k < 32; ++k) acc = fmaf(fv[k], W[k * 24 + d], acc);
    op[d] = fmaxf(acc, 0.f);
  }
}

// Head tail: out[n] = X[n,:] . w  (optionally softplus)
__global__ void head_dot_kernel(const float* __restrict__ X, const float* __restrict__ w,
                                int do_softplus, float* __restrict__ out, int N) {
  int n = blockIdx.x * blockDim.x + threadIdx.x;
  if (n >= N) return;
  const float* row = X + (size_t)n * 64;
  float s = 0.f;
#pragma unroll
  for (int k = 0; k < 64; ++k) s = fmaf(row[k], w[k], s);
  if (do_softplus) s = (s > 0.f) ? (s + log1pf(__expf(-s))) : log1pf(__expf(s));
  out[n] = s;
}

// ---------------- fused proj: LN(din) -> relu(GEMM din->64) -> relu(GEMM 64->64) ----------------
// 16 rows / block, 128 threads (4 waves, one 16-col tile each). dyn-LDS.
__global__ void __launch_bounds__(128) proj_kernel(
    const float* __restrict__ X, int DIN, int KPAD,
    const float* __restrict__ lng, const float* __restrict__ lnb,
    const unsigned short* __restrict__ W1p, const float* __restrict__ b1,
    const unsigned short* __restrict__ W2p, const float* __restrict__ b2,
    float* __restrict__ outf, unsigned short* __restrict__ outbf) {
  extern __shared__ char smem[];
  float* sx = (float*)smem;                                        // 16*KPAD f32
  unsigned short* sa = (unsigned short*)(sx + 16 * KPAD);          // 16*KPAD bf16
  unsigned short* sh = (unsigned short*)(sa + (size_t)16 * KPAD);  // 16*64 bf16
  float* smean = (float*)(sh + 16 * 64);
  float* sinv  = smean + 16;

  const int tid = threadIdx.x, lane = tid & 31, wv = tid >> 5;
  const int row0 = blockIdx.x * 16;

  for (int i = tid; i < 16 * DIN; i += 128) {
    int r = i / DIN, c = i - r * DIN;
    sx[r * KPAD + c] = X[(size_t)(row0 + r) * DIN + c];
  }
  __syncthreads();
  if (tid < 16) {
    float s = 0.f, s2 = 0.f;
    const float* rp = sx + tid * KPAD;
    for (int c = 0; c < DIN; ++c) { float v = rp[c]; s += v; s2 += v * v; }
    float mu = s / (float)DIN;
    float var = s2 / (float)DIN - mu * mu;
    smean[tid] = mu; sinv[tid] = rsqrtf(var + K_EPS);
  }
  __syncthreads();
  for (int i = tid; i < 16 * KPAD; i += 128) {
    int r = i / KPAD, c = i - r * KPAD;
    float v = 0.f;
    if (c < DIN) v = (sx[i] - smean[r]) * sinv[r] * lng[c] + lnb[c];
    sa[i] = f2bf(v);
  }
  __syncthreads();

  v8f_t acc = {0.f, 0.f, 0.f, 0.f, 0.f, 0.f, 0.f, 0.f};
  acc = gemm_tile(sa, KPAD, W1p, wv, KPAD >> 5, lane, acc);
  const int nc = wv * 16 + (lane & 15);
  const int kh = lane >> 4;
#pragma unroll
  for (int r = 0; r < 8; ++r) {
    int M = r + 8 * kh;
    sh[M * 64 + nc] = f2bf(fmaxf(acc[r] + b1[nc], 0.f));
  }
  __syncthreads();
  v8f_t acc2 = {0.f, 0.f, 0.f, 0.f, 0.f, 0.f, 0.f, 0.f};
  acc2 = gemm_tile(sh, 64, W2p, wv, 2, lane, acc2);
#pragma unroll
  for (int r = 0; r < 8; ++r) {
    int M = r + 8 * kh;
    float y = fmaxf(acc2[r] + b2[nc], 0.f);
    if (outf)  outf[(size_t)(row0 + M) * 64 + nc] = y;
    if (outbf) outbf[(size_t)(row0 + M) * 64 + nc] = f2bf(y);
  }
}

// ---------------- generic Y = act(X[N,64] @ W + b) ----------------
__global__ void __launch_bounds__(128) linear64_kernel(
    const float* __restrict__ X, const unsigned short* __restrict__ Wp,
    const float* __restrict__ bias, int dorelu, float* __restrict__ Y) {
  __shared__ __align__(16) unsigned short sa[16 * 64];
  const int tid = threadIdx.x, lane = tid & 31, wv = tid >> 5;
  const size_t row0 = (size_t)blockIdx.x * 16;
  for (int i = tid; i < 1024; i += 128) sa[i] = f2bf(X[row0 * 64 + i]);
  __syncthreads();
  v8f_t acc = {0.f, 0.f, 0.f, 0.f, 0.f, 0.f, 0.f, 0.f};
  acc = gemm_tile(sa, 64, Wp, wv, 2, lane, acc);
  const int nc = wv * 16 + (lane & 15);
  const int kh = lane >> 4;
#pragma unroll
  for (int r = 0; r < 8; ++r) {
    int M = r + 8 * kh;
    float y = acc[r] + (bias ? bias[nc] : 0.f);
    if (dorelu) y = fmaxf(y, 0.f);
    Y[(row0 + M) * 64 + nc] = y;
  }
}

// ---------------- fused edge message pipeline ----------------
// m = a[dst] + bm[src] + E0tile@We ; m2 = relu(LN(m))@Wf + bf ; agg[dst] += m2
__global__ void __launch_bounds__(128) edge_kernel(
    const unsigned short* __restrict__ E0,     // [E,64] bf16 row-major
    const int* __restrict__ dsti, const int* __restrict__ srci,
    const float* __restrict__ Ab, const float* __restrict__ Bm,
    const unsigned short* __restrict__ Wep,
    const float* __restrict__ lng, const float* __restrict__ lnb,
    const unsigned short* __restrict__ Wfp, const float* __restrict__ bfv,
    float* __restrict__ agg) {
  __shared__ float sm[16 * 64];
  __shared__ __align__(16) unsigned short sa[16 * 64];
  __shared__ float smean[16], sinv[16];
  __shared__ int sdst[16], ssrc[16];
  const int tid = threadIdx.x, lane = tid & 31, wv = tid >> 5;
  const int er0 = blockIdx.x * 16;
  if (tid < 16) { sdst[tid] = dsti[er0 + tid]; ssrc[tid] = srci[er0 + tid]; }
  __syncthreads();
  // gather a[dst]+bm[src] (L2-resident tables), coalesced into LDS
  for (int i = tid; i < 1024; i += 128) {
    int r = i >> 6, c = i & 63;
    sm[i] = Ab[(size_t)sdst[r] * 64 + c] + Bm[(size_t)ssrc[r] * 64 + c];
  }
  // E-tile @ We, A fragments straight from global bf16 (16B-aligned chunks)
  const int mr = lane & 15, kh = lane >> 4;
  v8f_t acc = {0.f, 0.f, 0.f, 0.f, 0.f, 0.f, 0.f, 0.f};
#pragma unroll
  for (int kb = 0; kb < 2; ++kb) {
    Frag a, b;
    const unsigned short* ap = E0 + (size_t)(er0 + mr) * 64 + kb * 32 + 8 * kh;
    *reinterpret_cast<uint4*>(&a.u[0]) = *reinterpret_cast<const uint4*>(ap);
    *reinterpret_cast<uint4*>(&a.u[4]) = *reinterpret_cast<const uint4*>(ap + 16);
    load_b_frag(b, Wep, kb, wv, lane);
    acc = wmma_bf16(a, b, acc);
  }
  __syncthreads();
  const int nc = wv * 16 + mr;
#pragma unroll
  for (int r = 0; r < 8; ++r) {
    int M = r + 8 * kh;
    sm[M * 64 + nc] = acc[r] + sm[M * 64 + nc];
  }
  __syncthreads();
  if (tid < 16) {
    float s = 0.f, s2 = 0.f;
    const float* rp = sm + tid * 64;
#pragma unroll
    for (int c = 0; c < 64; ++c) { float v = rp[c]; s += v; s2 += v * v; }
    float mu = s * (1.f / 64.f);
    float var = s2 * (1.f / 64.f) - mu * mu;
    smean[tid] = mu; sinv[tid] = rsqrtf(var + K_EPS);
  }
  __syncthreads();
  for (int i = tid; i < 1024; i += 128) {
    int r = i >> 6, c = i & 63;
    float v = (sm[i] - smean[r]) * sinv[r] * lng[c] + lnb[c];
    sa[i] = f2bf(fmaxf(v, 0.f));
  }
  __syncthreads();
  v8f_t acc2 = {0.f, 0.f, 0.f, 0.f, 0.f, 0.f, 0.f, 0.f};
  acc2 = gemm_tile(sa, 64, Wfp, wv, 2, lane, acc2);
#pragma unroll
  for (int r = 0; r < 8; ++r) {
    int M = r + 8 * kh;
    unsafeAtomicAdd(&agg[(size_t)sdst[M] * 64 + nc], acc2[r] + bfv[nc]);
  }
}

// ---------------- fused node update: LN(agg)||right -> relu(@Wo1+bo1) -> @Wo2+bo2 ----------------
__global__ void __launch_bounds__(128) node_update_kernel(
    const float* __restrict__ agg, const float* __restrict__ right,
    const float* __restrict__ lng, const float* __restrict__ lnb,
    const unsigned short* __restrict__ Wo1p, const float* __restrict__ bo1,
    const unsigned short* __restrict__ Wo2p, const float* __restrict__ bo2,
    float* __restrict__ out) {
  __shared__ float sx[16 * 64];
  __shared__ __align__(16) unsigned short sa[16 * 128];
  __shared__ __align__(16) unsigned short sh[16 * 64];
  __shared__ float smean[16], sinv[16];
  const int tid = threadIdx.x, lane = tid & 31, wv = tid >> 5;
  const int row0 = blockIdx.x * 16;
  for (int i = tid; i < 1024; i += 128) sx[i] = agg[(size_t)row0 * 64 + i];
  __syncthreads();
  if (tid < 16) {
    float s = 0.f, s2 = 0.f;
    const float* rp = sx + tid * 64;
#pragma unroll
    for (int c = 0; c < 64; ++c) { float v = rp[c]; s += v; s2 += v * v; }
    float mu = s * (1.f / 64.f);
    float var = s2 * (1.f / 64.f) - mu * mu;
    smean[tid] = mu; sinv[tid] = rsqrtf(var + K_EPS);
  }
  __syncthreads();
  for (int i = tid; i < 1024; i += 128) {
    int r = i >> 6, c = i & 63;
    sa[r * 128 + c]      = f2bf((sx[i] - smean[r]) * sinv[r] * lng[c] + lnb[c]);
    sa[r * 128 + 64 + c] = f2bf(right[(size_t)(row0 + r) * 64 + c]);
  }
  __syncthreads();
  v8f_t acc = {0.f, 0.f, 0.f, 0.f, 0.f, 0.f, 0.f, 0.f};
  acc = gemm_tile(sa, 128, Wo1p, wv, 4, lane, acc);
  const int nc = wv * 16 + (lane & 15);
  const int kh = lane >> 4;
#pragma unroll
  for (int r = 0; r < 8; ++r) {
    int M = r + 8 * kh;
    sh[M * 64 + nc] = f2bf(fmaxf(acc[r] + bo1[nc], 0.f));
  }
  __syncthreads();
  v8f_t acc2 = {0.f, 0.f, 0.f, 0.f, 0.f, 0.f, 0.f, 0.f};
  acc2 = gemm_tile(sh, 64, Wo2p, wv, 2, lane, acc2);
#pragma unroll
  for (int r = 0; r < 8; ++r) {
    int M = r + 8 * kh;
    out[(size_t)(row0 + M) * 64 + nc] = acc2[r] + bo2[nc];
  }
}

// ---------------- host ----------------
extern "C" void kernel_launch(void* const* d_in, const int* in_sizes, int n_in,
                              void* d_out, int out_size, void* d_ws, size_t ws_size,
                              hipStream_t stream) {
  (void)in_sizes; (void)out_size; (void)ws_size;
  if (n_in < 93) return;

  const float* cons_feat = (const float*)d_in[0];
  const int*   eidx      = (const int*)d_in[1];
  const float* edge_feat = (const float*)d_in[2];
  const float* var_feat  = (const float*)d_in[3];
  const int* ci = eidx;              // row 0: constraint indices
  const int* vi = eidx + K_NEDGES;   // row 1: variable indices

  // params: top-level sorted keys, nested dicts sorted (jax tree-flatten order)
  int p = 4;
  auto nx = [&]() { return (const float*)d_in[p++]; };
  struct EmbP  { const float *W, *b, *coef; };
  struct HeadP { const float *W1, *W2, *b1; };
  struct ProjP { const float *W1, *W2, *b1, *b2, *ln_b, *ln_g; };
  struct ConvP { const float *We, *Wf, *Wl, *Wo1, *Wo2, *Wr, *bf, *bl, *bo1, *bo2,
                             *lnf_b, *lnf_g, *lnp_b, *lnp_g; };
  auto rdEmb  = [&]() { EmbP e;  e.W = nx(); e.b = nx(); e.coef = nx(); return e; };
  auto rdHead = [&]() { HeadP h; h.W1 = nx(); h.W2 = nx(); h.b1 = nx(); return h; };
  auto rdProj = [&]() { ProjP q; q.W1 = nx(); q.W2 = nx(); q.b1 = nx(); q.b2 = nx();
                        q.ln_b = nx(); q.ln_g = nx(); return q; };
  auto rdConv = [&]() { ConvP c; c.We = nx(); c.Wf = nx(); c.Wl = nx(); c.Wo1 = nx();
                        c.Wo2 = nx(); c.Wr = nx(); c.bf = nx(); c.bl = nx();
                        c.bo1 = nx(); c.bo2 = nx(); c.lnf_b = nx(); c.lnf_g = nx();
                        c.lnp_b = nx(); c.lnp_g = nx(); return c; };

  EmbP  cons_emb  = rdEmb();
  HeadP cons_head = rdHead();
  ProjP cons_proj = rdProj();
  ConvP conv_c_to_v  = rdConv();
  ConvP conv_c_to_v2 = rdConv();
  ConvP conv_v_to_c  = rdConv();
  ConvP conv_v_to_c2 = rdConv();
  EmbP  edge_emb  = rdEmb();
  ProjP edge_proj = rdProj();
  EmbP  var_emb   = rdEmb();
  HeadP var_head  = rdHead();
  ProjP var_proj  = rdProj();

  // workspace bump allocator
  char* base = (char*)d_ws; size_t off = 0;
  auto alloc = [&](size_t bytes) -> void* {
    void* r = base + off; off = (off + bytes + 255) & ~(size_t)255; return r;
  };
  float* EMB  = (float*)alloc((size_t)K_NVARS * 432 * 4);          // embed scratch (max)
  unsigned short* E0 = (unsigned short*)alloc((size_t)K_NEDGES * 64 * 2);  // bf16 edge feats
  float* C0   = (float*)alloc((size_t)K_NCONS * 64 * 4);
  float* C1   = (float*)alloc((size_t)K_NCONS * 64 * 4);
  float* V0   = (float*)alloc((size_t)K_NVARS * 64 * 4);
  float* V1   = (float*)alloc((size_t)K_NVARS * 64 * 4);
  float* ABUF = (float*)alloc((size_t)K_NVARS * 64 * 4);
  float* BMBUF= (float*)alloc((size_t)K_NVARS * 64 * 4);
  float* AGG  = (float*)alloc((size_t)K_NVARS * 64 * 4);
  float* TMP  = (float*)alloc((size_t)K_NVARS * 64 * 4);

  auto packW = [&](const float* W, int K, int Kpad) -> const unsigned short* {
    unsigned short* P = (unsigned short*)alloc((size_t)Kpad * 64 * 2);
    int total = (Kpad >> 5) * 2048;
    pack_w_kernel<<<(total + 255) / 256, 256, 0, stream>>>(W, K, Kpad, P);
    return P;
  };
  struct ConvPk { const unsigned short *We, *Wf, *Wl, *Wo1, *Wo2, *Wr; };
  auto packConv = [&](const ConvP& c) {
    ConvPk k;
    k.We = packW(c.We, 64, 64);   k.Wf = packW(c.Wf, 64, 64);
    k.Wl = packW(c.Wl, 64, 64);   k.Wo1 = packW(c.Wo1, 128, 128);
    k.Wo2 = packW(c.Wo2, 64, 64); k.Wr = packW(c.Wr, 64, 64);
    return k;
  };

  const unsigned short* consW1p = packW(cons_proj.W1, 96, 96);
  const unsigned short* consW2p = packW(cons_proj.W2, 64, 64);
  const unsigned short* varW1p  = packW(var_proj.W1, 432, 448);
  const unsigned short* varW2p  = packW(var_proj.W2, 64, 64);
  const unsigned short* edgeW1p = packW(edge_proj.W1, 24, 32);
  const unsigned short* edgeW2p = packW(edge_proj.W2, 64, 64);
  ConvPk pk_vc  = packConv(conv_v_to_c);
  ConvPk pk_cv  = packConv(conv_c_to_v);
  ConvPk pk_vc2 = packConv(conv_v_to_c2);
  ConvPk pk_cv2 = packConv(conv_c_to_v2);
  const unsigned short* varHW1p  = packW(var_head.W1, 64, 64);
  const unsigned short* consHW1p = packW(cons_head.W1, 64, 64);

  auto smemProj = [](int KPAD) { return (size_t)96 * KPAD + 2176; };

  // ---- embeddings + projections ----
  embed_kernel<<<(K_NCONS * 4 + 255) / 256, 256, 0, stream>>>(
      cons_feat, K_NCONS, 4, cons_emb.coef, cons_emb.W, cons_emb.b, EMB);
  proj_kernel<<<K_NCONS / 16, 128, smemProj(96), stream>>>(
      EMB, 96, 96, cons_proj.ln_g, cons_proj.ln_b, consW1p, cons_proj.b1,
      consW2p, cons_proj.b2, C0, (unsigned short*)nullptr);

  embed_kernel<<<(K_NVARS * 18 + 255) / 256, 256, 0, stream>>>(
      var_feat, K_NVARS, 18, var_emb.coef, var_emb.W, var_emb.b, EMB);
  proj_kernel<<<K_NVARS / 16, 128, smemProj(448), stream>>>(
      EMB, 432, 448, var_proj.ln_g, var_proj.ln_b, varW1p, var_proj.b1,
      varW2p, var_proj.b2, V0, (unsigned short*)nullptr);

  embed_kernel<<<(K_NEDGES * 1 + 255) / 256, 256, 0, stream>>>(
      edge_feat, K_NEDGES, 1, edge_emb.coef, edge_emb.W, edge_emb.b, EMB);
  proj_kernel<<<K_NEDGES / 16, 128, smemProj(32), stream>>>(
      EMB, 24, 32, edge_proj.ln_g, edge_proj.ln_b, edgeW1p, edge_proj.b1,
      edgeW2p, edge_proj.b2, (float*)nullptr, E0);   // store edge feats as bf16

  // ---- bipartite conv layers ----
  auto run_conv = [&](const ConvP& c, const ConvPk& k,
                      const float* left, int Nl, const int* src, const int* dst,
                      const float* right, int Nr, float* outp) {
    linear64_kernel<<<Nr / 16, 128, 0, stream>>>(right, k.Wl, c.bl, 0, ABUF);
    linear64_kernel<<<Nl / 16, 128, 0, stream>>>(left, k.Wr, (const float*)nullptr, 0, BMBUF);
    long long n = (long long)Nr * 64;
    zero_kernel<<<(int)((n + 255) / 256), 256, 0, stream>>>(AGG, n);
    edge_kernel<<<K_NEDGES / 16, 128, 0, stream>>>(
        E0, dst, src, ABUF, BMBUF, k.We, c.lnf_g, c.lnf_b, k.Wf, c.bf, AGG);
    node_update_kernel<<<Nr / 16, 128, 0, stream>>>(
        AGG, right, c.lnp_g, c.lnp_b, k.Wo1, c.bo1, k.Wo2, c.bo2, outp);
  };

  run_conv(conv_v_to_c,  pk_vc,  V0, K_NVARS, vi, ci, C0, K_NCONS, C1);
  run_conv(conv_c_to_v,  pk_cv,  C1, K_NCONS, ci, vi, V0, K_NVARS, V1);
  run_conv(conv_v_to_c2, pk_vc2, V1, K_NVARS, vi, ci, C1, K_NCONS, C0);
  run_conv(conv_c_to_v2, pk_cv2, C0, K_NCONS, ci, vi, V1, K_NVARS, V0);

  // ---- heads: x_all then softplus(lam_all), concatenated in d_out ----
  float* out = (float*)d_out;
  linear64_kernel<<<K_NVARS / 16, 128, 0, stream>>>(V0, varHW1p, var_head.b1, 1, TMP);
  head_dot_kernel<<<(K_NVARS + 127) / 128, 128, 0, stream>>>(TMP, var_head.W2, 0, out, K_NVARS);
  linear64_kernel<<<K_NCONS / 16, 128, 0, stream>>>(C0, consHW1p, cons_head.b1, 1, TMP);
  head_dot_kernel<<<(K_NCONS + 127) / 128, 128, 0, stream>>>(TMP, cons_head.W2, 1, out + K_NVARS, K_NCONS);
}
